// VectorQuantizer_25220047962174
// MI455X (gfx1250) — matrix operand order repaired and checked by reading, not compiled, not run on hardware
//
#include <hip/hip_runtime.h>

// ---------------------------------------------------------------------------
// VQ-VAE vector quantizer for MI455X (gfx1250, wave32, WMMA).
//   latents  : [64, 64, 32, 32] f32  (B, D, H, W)
//   codebook : [1024, 64] f32
//   out      : [64, 64, 32, 32] f32 quantized (+ 1 scalar vq_loss at the end)
// Distance argmin via bf16-split GEMM on v_wmma_f32_16x16x32_bf16.
// ---------------------------------------------------------------------------

typedef __attribute__((ext_vector_type(16))) __bf16 v16bf;
typedef __attribute__((ext_vector_type(8)))  float  v8f;

#define KCODES  1024
#define DIM     64
#define ROWS_PB 128          // rows (hw positions) per block
#define LDSTR   72           // 64 + 8 pad: rows stay 16B aligned, banks spread

// -------------------- prologue: codebook split + norms ---------------------
__global__ __launch_bounds__(256) void vq_prep(const float* __restrict__ codebook,
                                               __bf16* __restrict__ cb_hi,
                                               __bf16* __restrict__ cb_lo,
                                               float*  __restrict__ cnorm) {
    int k = blockIdx.x * blockDim.x + threadIdx.x;
    if (k >= KCODES) return;
    float acc = 0.f;
    for (int d = 0; d < DIM; ++d) {
        float c  = codebook[k * DIM + d];
        acc     += c * c;
        __bf16 h = (__bf16)c;
        __bf16 l = (__bf16)(c - (float)h);
        cb_hi[k * DIM + d] = h;
        cb_lo[k * DIM + d] = l;
    }
    cnorm[k] = acc;
}

// -------------------- fragment loaders -------------------------------------
// A (16x32 bf16, MxK): lane<16 -> M=lane,    K = {0..7, 16..23} (+chunk)
//                      lane>=16 -> M=lane-16, K = {8..15, 24..31} (+chunk)
__device__ __forceinline__ v16bf load_a(const __bf16 (*m)[LDSTR], int row,
                                        int d0, int d1) {
    union { uint4 u[2]; v16bf v; } f;
    f.u[0] = *(const uint4*)&m[row][d0];
    f.u[1] = *(const uint4*)&m[row][d1];
    return f.v;
}

// B (32x16 bf16, KxN): lane -> N = lane&15 ; K = (lane<16 ? {0..7,16..23} : {8..15,24..31})
__device__ __forceinline__ v16bf load_b(const __bf16* __restrict__ cb,
                                        int code, int d0) {
    union { uint4 u[2]; v16bf v; } f;
    f.u[0] = *(const uint4*)(cb + code * DIM + d0);
    f.u[1] = *(const uint4*)(cb + code * DIM + d0 + 16);
    return f.v;
}

// -------------------- main kernel ------------------------------------------
__global__ __launch_bounds__(256) void vq_main(const float* __restrict__ latents,
                                               const float* __restrict__ codebook,
                                               const __bf16* __restrict__ cb_hi,
                                               const __bf16* __restrict__ cb_lo,
                                               const float*  __restrict__ cnorm,
                                               float* __restrict__ out,
                                               float* __restrict__ partials) {
    __shared__ __bf16 xhi[ROWS_PB][LDSTR];
    __shared__ __bf16 xlo[ROWS_PB][LDSTR];
    __shared__ int    widx[ROWS_PB];
    __shared__ float  wsum[8];

    const int t   = threadIdx.x;
    const int b   = blockIdx.x >> 3;          // image index
    const int hw0 = (blockIdx.x & 7) << 7;    // 128-row window inside image
    const float* xbase = latents + (size_t)b * (DIM * 1024) + hw0;

    // ---- load X tile (coalesced along hw), split into bf16 hi/lo ----
    for (int e = t; e < ROWS_PB * DIM; e += 256) {
        int d = e >> 7, r = e & 127;
        float  x = xbase[d * 1024 + r];
        __bf16 h = (__bf16)x;
        __bf16 l = (__bf16)(x - (float)h);
        xhi[r][d] = h;
        xlo[r][d] = l;
    }
    __syncthreads();

    const int lane  = t & 31;
    const int wave  = t >> 5;
    const int rbase = wave * 16;              // 16 rows per wave
    const int mrow  = rbase + (lane & 15);
    const int kb    = (lane < 16) ? 0 : 8;    // A-fragment K sub-offset
    const int bd0   = (lane < 16) ? 0 : 8;    // B-fragment K sub-offset

    // A fragments are loop-invariant: hoist (4 x 8 VGPRs)
    v16bf Ah0 = load_a(xhi, mrow, kb,      kb + 16);
    v16bf Ah1 = load_a(xhi, mrow, 32 + kb, 32 + kb + 16);
    v16bf Al0 = load_a(xlo, mrow, kb,      kb + 16);
    v16bf Al1 = load_a(xlo, mrow, 32 + kb, 32 + kb + 16);

    // running argmax of (x.c - 0.5*||c||^2)  ==  argmin distance
    float best[8];
    int   bidx[8];
#pragma unroll
    for (int i = 0; i < 8; ++i) { best[i] = -3.0e38f; bidx[i] = 0; }

    for (int kt = 0; kt < KCODES / 16; ++kt) {
        const int code = kt * 16 + (lane & 15);     // this lane's column/code
        const float init = -0.5f * cnorm[code];

        v16bf Bh0 = load_b(cb_hi, code, bd0);
        v16bf Bh1 = load_b(cb_hi, code, 32 + bd0);
        v16bf Bl0 = load_b(cb_lo, code, bd0);
        v16bf Bl1 = load_b(cb_lo, code, 32 + bd0);

        v8f c = { init, init, init, init, init, init, init, init };
        // hi*hi (both 32-wide K chunks), then cross terms hi*lo + lo*hi
        c = __builtin_amdgcn_wmma_f32_16x16x32_bf16(false, Ah0, false, Bh0, (short)0, c, false, false);
        c = __builtin_amdgcn_wmma_f32_16x16x32_bf16(false, Ah1, false, Bh1, (short)0, c, false, false);
        c = __builtin_amdgcn_wmma_f32_16x16x32_bf16(false, Ah0, false, Bl0, (short)0, c, false, false);
        c = __builtin_amdgcn_wmma_f32_16x16x32_bf16(false, Ah1, false, Bl1, (short)0, c, false, false);
        c = __builtin_amdgcn_wmma_f32_16x16x32_bf16(false, Al0, false, Bh0, (short)0, c, false, false);
        c = __builtin_amdgcn_wmma_f32_16x16x32_bf16(false, Al1, false, Bh1, (short)0, c, false, false);

#pragma unroll
        for (int i = 0; i < 8; ++i) {
            if (c[i] > best[i]) { best[i] = c[i]; bidx[i] = code; }   // strict > keeps first idx
        }
    }

    // C/D layout: vgpr i, lanes 0-15 -> row rbase+i, lanes 16-31 -> rbase+8+i.
    // Reduce the 16 columns held across each 16-lane half (xor masks < 16).
#pragma unroll
    for (int i = 0; i < 8; ++i) {
        float s  = best[i];
        int   id = bidx[i];
#pragma unroll
        for (int off = 1; off < 16; off <<= 1) {
            float os = __shfl_xor(s,  off, 32);
            int   oi = __shfl_xor(id, off, 32);
            if (os > s || (os == s && oi < id)) { s = os; id = oi; }
        }
        bidx[i] = id;
    }
    if (lane == 0) {
#pragma unroll
        for (int i = 0; i < 8; ++i) widx[rbase + i] = bidx[i];
    } else if (lane == 16) {
#pragma unroll
        for (int i = 0; i < 8; ++i) widx[rbase + 8 + i] = bidx[i];
    }
    __syncthreads();

    // ---- gather winning codes -> out ([B,D,H,W]), fused loss MSE ----
    float acc = 0.f;
    float* obase = out + (size_t)b * (DIM * 1024) + hw0;
    for (int e = t; e < ROWS_PB * DIM; e += 256) {
        int d = e >> 7, r = e & 127;
        float q  = codebook[widx[r] * DIM + d];
        float x  = xbase[d * 1024 + r];
        obase[d * 1024 + r] = q;                  // coalesced along hw
        float dq = q - x;
        acc += dq * dq;
    }
#pragma unroll
    for (int off = 16; off; off >>= 1) acc += __shfl_xor(acc, off, 32);
    if (lane == 0) wsum[wave] = acc;
    __syncthreads();
    if (t == 0) {
        float s = 0.f;
        for (int i = 0; i < 8; ++i) s += wsum[i];
        partials[blockIdx.x] = s;                 // deterministic 2-pass loss
    }
}

// -------------------- final loss reduce (deterministic) --------------------
__global__ __launch_bounds__(256) void vq_final(const float* __restrict__ partials,
                                                int n, float scale,
                                                float* __restrict__ loss_out) {
    __shared__ float sh[256];
    float a = 0.f;
    for (int i = threadIdx.x; i < n; i += 256) a += partials[i];
    sh[threadIdx.x] = a;
    __syncthreads();
    for (int s = 128; s; s >>= 1) {
        if (threadIdx.x < s) sh[threadIdx.x] += sh[threadIdx.x + s];
        __syncthreads();
    }
    if (threadIdx.x == 0) *loss_out = sh[0] * scale;
}

// ---------------------------------------------------------------------------
extern "C" void kernel_launch(void* const* d_in, const int* in_sizes, int n_in,
                              void* d_out, int out_size, void* d_ws, size_t ws_size,
                              hipStream_t stream) {
    const float* latents  = (const float*)d_in[0];
    const float* codebook = (const float*)d_in[1];
    float* out = (float*)d_out;

    // workspace layout: cb_hi[64K bf16] | cb_lo[64K bf16] | cnorm[1K f32] | partials
    __bf16* cb_hi    = (__bf16*)d_ws;
    __bf16* cb_lo    = cb_hi + (size_t)KCODES * DIM;
    float*  cnorm    = (float*)(cb_lo + (size_t)KCODES * DIM);
    float*  partials = cnorm + KCODES;

    const int n_elems  = in_sizes[0];                     // B*D*H*W = 4194304
    const int nblocks  = n_elems / (ROWS_PB * DIM);       // 512
    float*    loss_out = out + (out_size - 1);
    const float scale  = 1.25f / (float)(out_size - 1);   // (1+beta)/mean-count

    vq_prep <<<KCODES / 256, 256, 0, stream>>>(codebook, cb_hi, cb_lo, cnorm);
    vq_main <<<nblocks,      256, 0, stream>>>(latents, codebook, cb_hi, cb_lo,
                                               cnorm, out, partials);
    vq_final<<<1,            256, 0, stream>>>(partials, nblocks, scale, loss_out);
}
